// Block_15710990369356
// MI455X (gfx1250) — matrix operand order repaired and checked by reading, not compiled, not run on hardware
//
#include <hip/hip_runtime.h>
#include <math.h>
#include <stdint.h>

typedef __attribute__((ext_vector_type(16))) _Float16 v16h;
typedef __attribute__((ext_vector_type(8)))  _Float16 v8h;
typedef __attribute__((ext_vector_type(8)))  float    v8f;

#define DIMC    1024
#define HEADS   16
#define HEAD_D  64
#define HIDDEN  4096
#define BATCH   4
#define SEQ     2048
#define MTOT    (BATCH * SEQ)

// LDS tile pitch in halves: 40 (80B) -> the 16 lanes of a fragment read start
// at banks 20*r mod 64 = all multiples of 4, so ds_load_b128 is conflict-free.
#define APITCH  40

// ---------------------------------------------------------------------------
// WMMA fragment loaders (wave32; K order is this code's contiguous choice, same for A and B)
// A 16x32 f16: lane holds row (lane&15); K = hf*16+{0..15} contiguous
// B 32x16 f16: lane holds col (lane&15); K = hf*16+{0..15} contiguous
// ---------------------------------------------------------------------------
__device__ __forceinline__ v16h load_a_frag(const _Float16* base, int row0,
                                            int ld, int k0, int lane) {
  const int r  = row0 + (lane & 15);
  const int hf = (lane >> 4) * 16;
  const _Float16* p = base + (size_t)r * ld + (k0 + hf);
  union { v8h h[2]; v16h v; } u;
  u.h[0] = *(const v8h*)(p);
  u.h[1] = *(const v8h*)(p + 8);
  return u.v;
}

__device__ __forceinline__ v16h load_b_frag(const _Float16* base, int col0,
                                            int ld, int k0, int lane) {
  const int c  = col0 + (lane & 15);
  const int hf = (lane >> 4) * 16;
  return *(const v16h*)(base + (size_t)c * ld + (k0 + hf));
}

__device__ __forceinline__ v8f wmma_f16(v16h a, v16h b, v8f c) {
  return __builtin_amdgcn_wmma_f32_16x16x32_f16(false, a, false, b,
                                                (short)0, c, false, false);
}

// Async copy 16B global -> LDS (gfx1250, ASYNCcnt-tracked)
__device__ __forceinline__ void async_b128(const _Float16* gptr, void* lptr) {
  uint32_t lds_off = (uint32_t)(uintptr_t)lptr;   // low 32b of LDS aperture addr
  uint64_t gaddr   = (uint64_t)(uintptr_t)gptr;
  asm volatile("global_load_async_to_lds_b128 %0, %1, off"
               :: "v"(lds_off), "v"(gaddr) : "memory");
}

__device__ __forceinline__ void wait_async0() {
  asm volatile("s_wait_asynccnt 0" ::: "memory");
}

// ---------------------------------------------------------------------------
// Weight cast + transpose: W[K,N] f32 -> Wt[N,K] f16 (columns contiguous)
// ---------------------------------------------------------------------------
__global__ void cast_transpose(const float* __restrict__ in,
                               _Float16* __restrict__ out, int K, int N) {
  size_t idx = (size_t)blockIdx.x * 256 + threadIdx.x;
  if (idx >= (size_t)K * N) return;
  int kk = (int)(idx / N), n = (int)(idx % N);
  out[(size_t)n * K + kk] = (_Float16)in[idx];
}

// ---------------------------------------------------------------------------
// LayerNorm (row per block) fused with f32 -> f16 cast
// ---------------------------------------------------------------------------
__global__ __launch_bounds__(256) void ln_kernel(const float* __restrict__ x,
                                                 const float* __restrict__ w,
                                                 const float* __restrict__ b,
                                                 _Float16* __restrict__ out) {
  __shared__ float red[256];
  const int row = blockIdx.x;
  const int t = threadIdx.x;
  const float* xr = x + (size_t)row * DIMC;
  float loc[4];
  float s = 0.f;
#pragma unroll
  for (int i = 0; i < 4; ++i) { loc[i] = xr[t + i * 256]; s += loc[i]; }
  red[t] = s; __syncthreads();
  for (int off = 128; off > 0; off >>= 1) {
    if (t < off) red[t] += red[t + off];
    __syncthreads();
  }
  const float mean = red[0] * (1.0f / DIMC);
  __syncthreads();
  float sq = 0.f;
#pragma unroll
  for (int i = 0; i < 4; ++i) { float d = loc[i] - mean; sq += d * d; }
  red[t] = sq; __syncthreads();
  for (int off = 128; off > 0; off >>= 1) {
    if (t < off) red[t] += red[t + off];
    __syncthreads();
  }
  const float rstd = rsqrtf(red[0] * (1.0f / DIMC) + 1e-6f);
  _Float16* orow = out + (size_t)row * DIMC;
#pragma unroll
  for (int i = 0; i < 4; ++i) {
    int idx = t + i * 256;
    orow[idx] = (_Float16)(w[idx] * (loc[i] - mean) * rstd + b[idx]);
  }
}

// ---------------------------------------------------------------------------
// WMMA GEMM with double-buffered async-LDS staging.
// C[M,N] = A[M,K](f16,row-major) * Bt[N,K](f16, columns of B contiguous)
// Block: 256 thr = 8 waves; block tile 128x128; wave tile 64x32 (4x2);
// K-step 32. Each K-step: 16KB of operands copied once per block via
// GLOBAL_LOAD_ASYNC_TO_LDS_B128 (overlapped with WMMA of previous step).
// MODE 0: scatter to Q/K/Vt head layouts (+bias)
// MODE 1: fp32 out = acc + bias + residual
// MODE 2: f16 out = gelu_exact(acc + bias)
// ---------------------------------------------------------------------------
template <int MODE>
__global__ __launch_bounds__(256) void gemm_wmma(
    const _Float16* __restrict__ A, const _Float16* __restrict__ Bt,
    const float* __restrict__ bias, int M, int N, int K,
    _Float16* __restrict__ outH, float* __restrict__ outF,
    const float* __restrict__ residual,
    _Float16* __restrict__ qO, _Float16* __restrict__ kO,
    _Float16* __restrict__ vtO) {
  __shared__ _Float16 lds_a[2][128][APITCH];
  __shared__ _Float16 lds_b[2][128][APITCH];

  const int tid   = threadIdx.x;
  const int lane  = tid & 31;
  const int wave  = tid >> 5;
  const int waveM = wave >> 2, waveN = wave & 3;
  const int rowBlk = blockIdx.y * 128;
  const int colBlk = blockIdx.x * 128;

  // Stage one 128x32 A tile + one 128x32 B tile: 512 chunks of 16B each,
  // 2 chunks per thread per matrix.
  auto issue_tile = [&](int k0, int bi) {
#pragma unroll
    for (int c = 0; c < 2; ++c) {
      const int chunk = tid * 2 + c;
      const int r = chunk >> 2, part = (chunk & 3) * 8;   // 8 halves = 16B
      async_b128(A + (size_t)(rowBlk + r) * K + k0 + part,
                 &lds_a[bi][r][part]);
      async_b128(Bt + (size_t)(colBlk + r) * K + k0 + part,
                 &lds_b[bi][r][part]);
    }
  };

  v8f zero = {};
  v8f acc[4][2];
#pragma unroll
  for (int i = 0; i < 4; ++i)
#pragma unroll
    for (int j = 0; j < 2; ++j) acc[i][j] = zero;

  const int nk = K / 32;
  issue_tile(0, 0);

  for (int kt = 0; kt < nk; ++kt) {
    wait_async0();        // this wave's copies for tile kt have landed
    __syncthreads();      // everyone's copies for tile kt are visible
    if (kt + 1 < nk) issue_tile((kt + 1) * 32, (kt + 1) & 1);

    const int bi = kt & 1;
    const int hf8  = (lane >> 4) * 8;
    const int hf16 = (lane >> 4) * 16;
    v16h a[4], b[2];
#pragma unroll
    for (int i = 0; i < 4; ++i) {
      const _Float16* p = &lds_a[bi][waveM * 64 + i * 16 + (lane & 15)][hf16];
      union { v8h h[2]; v16h v; } u;
      u.h[0] = *(const v8h*)(p);
      u.h[1] = *(const v8h*)(p + 8);
      a[i] = u.v;
    }
#pragma unroll
    for (int j = 0; j < 2; ++j)
      b[j] = *(const v16h*)&lds_b[bi][waveN * 32 + j * 16 + (lane & 15)][hf16];

#pragma unroll
    for (int i = 0; i < 4; ++i)
#pragma unroll
      for (int j = 0; j < 2; ++j)
        acc[i][j] = wmma_f16(a[i], b[j], acc[i][j]);

    __syncthreads();      // all reads of buf bi done before it is re-filled
  }

  const int half = lane >> 4, nn = lane & 15;
#pragma unroll
  for (int i = 0; i < 4; ++i) {
#pragma unroll
    for (int j = 0; j < 2; ++j) {
      const int cc = colBlk + waveN * 32 + j * 16 + nn;
      const float bv = bias[cc];
#pragma unroll
      for (int r = 0; r < 8; ++r) {
        const int rr = rowBlk + waveM * 64 + i * 16 + half * 8 + r;
        float v = acc[i][j][r] + bv;
        if (MODE == 0) {
          int which = cc >> 10;
          int c1 = cc & 1023;
          int hh = c1 >> 6, d = c1 & 63;
          int bb = rr >> 11, nrow = rr & 2047;
          size_t bhz = (size_t)(bb * HEADS + hh);
          if (which == 0)
            qO[(bhz * SEQ + nrow) * HEAD_D + d] = (_Float16)v;
          else if (which == 1)
            kO[(bhz * SEQ + nrow) * HEAD_D + d] = (_Float16)v;
          else
            vtO[(bhz * HEAD_D + d) * SEQ + nrow] = (_Float16)v;
        } else if (MODE == 1) {
          size_t idx = (size_t)rr * N + cc;
          outF[idx] = v + residual[idx];
        } else {
          float g = 0.5f * v * (1.0f + erff(v * 0.70710678118654752f));
          outH[(size_t)rr * N + cc] = (_Float16)g;
        }
      }
    }
  }
}

// ---------------------------------------------------------------------------
// Flash-style attention. Wave = 16 query rows; stream key blocks of 32.
// S = Q Kt (4 WMMA), online softmax (shfl_xor row reductions), P staged
// through LDS into A-fragment layout, O += P V (4 WMMA vs pre-transposed Vt).
// ---------------------------------------------------------------------------
__global__ __launch_bounds__(128) void attn_kernel(
    const _Float16* __restrict__ Q, const _Float16* __restrict__ Km,
    const _Float16* __restrict__ Vt, _Float16* __restrict__ out) {
  __shared__ _Float16 pstage[4][16][32];
  const int lane = threadIdx.x & 31;
  const int wave = threadIdx.x >> 5;
  const int bh = blockIdx.x >> 5;      // 0..63 (b*16 + h)
  const int chunk = blockIdx.x & 31;   // 32 chunks of 64 rows
  const int bb = bh >> 4, hh = bh & 15;
  const _Float16* q  = Q  + (size_t)bh * SEQ * HEAD_D;
  const _Float16* kp = Km + (size_t)bh * SEQ * HEAD_D;
  const _Float16* vt = Vt + (size_t)bh * HEAD_D * SEQ;
  const int q0 = chunk * 64 + wave * 16;
  const int half = lane >> 4, nn = lane & 15;

  v8f zero = {};
  v8f o[4];
#pragma unroll
  for (int dt = 0; dt < 4; ++dt) o[dt] = zero;
  float mrun[8], lrun[8];
#pragma unroll
  for (int r = 0; r < 8; ++r) { mrun[r] = -1e30f; lrun[r] = 0.f; }

  for (int kb = 0; kb < SEQ; kb += 32) {
    v8f s0 = zero, s1 = zero;
#pragma unroll
    for (int d0 = 0; d0 < 64; d0 += 32) {
      v16h aq = load_a_frag(q, q0, HEAD_D, d0, lane);
      v16h b0 = load_b_frag(kp, kb, HEAD_D, d0, lane);
      v16h b1 = load_b_frag(kp, kb + 16, HEAD_D, d0, lane);
      s0 = wmma_f16(aq, b0, s0);
      s1 = wmma_f16(aq, b1, s1);
    }
    float p0[8], p1[8], mx[8];
#pragma unroll
    for (int r = 0; r < 8; ++r) {
      p0[r] = s0[r] * 0.125f;   // HEAD_D^-0.5
      p1[r] = s1[r] * 0.125f;
      mx[r] = fmaxf(p0[r], p1[r]);
    }
#pragma unroll
    for (int m = 1; m < 16; m <<= 1)
#pragma unroll
      for (int r = 0; r < 8; ++r)
        mx[r] = fmaxf(mx[r], __shfl_xor(mx[r], m, 32));
    float corr[8], su[8];
#pragma unroll
    for (int r = 0; r < 8; ++r) {
      float mnew = fmaxf(mrun[r], mx[r]);
      corr[r] = __expf(mrun[r] - mnew);
      mrun[r] = mnew;
      p0[r] = __expf(p0[r] - mnew);
      p1[r] = __expf(p1[r] - mnew);
      su[r] = p0[r] + p1[r];
    }
#pragma unroll
    for (int m = 1; m < 16; m <<= 1)
#pragma unroll
      for (int r = 0; r < 8; ++r) su[r] += __shfl_xor(su[r], m, 32);
#pragma unroll
    for (int r = 0; r < 8; ++r) lrun[r] = lrun[r] * corr[r] + su[r];
#pragma unroll
    for (int dt = 0; dt < 4; ++dt)
#pragma unroll
      for (int r = 0; r < 8; ++r) o[dt][r] *= corr[r];

    // Transpose P from C-layout to A-fragment layout via LDS
#pragma unroll
    for (int r = 0; r < 8; ++r) {
      pstage[wave][half * 8 + r][nn]      = (_Float16)p0[r];
      pstage[wave][half * 8 + r][16 + nn] = (_Float16)p1[r];
    }
    __syncthreads();
    union { v8h h[2]; v16h v; } up;
    up.h[0] = *(const v8h*)&pstage[wave][nn][half * 16];
    up.h[1] = *(const v8h*)&pstage[wave][nn][half * 16 + 8];
    v16h pf = up.v;
    __syncthreads();

#pragma unroll
    for (int dt = 0; dt < 4; ++dt) {
      v16h vb = load_b_frag(vt, dt * 16, SEQ, kb, lane);
      o[dt] = wmma_f16(pf, vb, o[dt]);
    }
  }

#pragma unroll
  for (int dt = 0; dt < 4; ++dt)
#pragma unroll
    for (int r = 0; r < 8; ++r) {
      int nrow = q0 + half * 8 + r;
      int col = hh * HEAD_D + dt * 16 + nn;
      float val = o[dt][r] / lrun[r];
      out[((size_t)(bb * SEQ + nrow)) * DIMC + col] = (_Float16)val;
    }
}

// ---------------------------------------------------------------------------
extern "C" void kernel_launch(void* const* d_in, const int* in_sizes, int n_in,
                              void* d_out, int out_size, void* d_ws,
                              size_t ws_size, hipStream_t stream) {
  const float* x      = (const float*)d_in[0];
  const float* ln1_w  = (const float*)d_in[1];
  const float* ln1_b  = (const float*)d_in[2];
  const float* qkv_w  = (const float*)d_in[3];
  const float* qkv_b  = (const float*)d_in[4];
  const float* proj_w = (const float*)d_in[5];
  const float* proj_b = (const float*)d_in[6];
  const float* ln2_w  = (const float*)d_in[7];
  const float* ln2_b  = (const float*)d_in[8];
  const float* fc1_w  = (const float*)d_in[9];
  const float* fc1_b  = (const float*)d_in[10];
  const float* fc2_w  = (const float*)d_in[11];
  const float* fc2_b  = (const float*)d_in[12];
  float* outp = (float*)d_out;

  char* ws = (char*)d_ws;
  size_t off = 0;
  auto alloc = [&](size_t bytes) -> void* {
    void* p = ws + off;
    off += (bytes + 255) & ~(size_t)255;
    return p;
  };
  _Float16* xh     = (_Float16*)alloc((size_t)MTOT * DIMC * 2);
  _Float16* wqkvT  = (_Float16*)alloc((size_t)3 * DIMC * DIMC * 2);
  _Float16* wprojT = (_Float16*)alloc((size_t)DIMC * DIMC * 2);
  _Float16* wfc1T  = (_Float16*)alloc((size_t)HIDDEN * DIMC * 2);
  _Float16* wfc2T  = (_Float16*)alloc((size_t)DIMC * HIDDEN * 2);
  _Float16* Qb     = (_Float16*)alloc((size_t)MTOT * DIMC * 2);
  _Float16* Kb     = (_Float16*)alloc((size_t)MTOT * DIMC * 2);
  _Float16* Vtb    = (_Float16*)alloc((size_t)MTOT * DIMC * 2);
  _Float16* attn_o = (_Float16*)alloc((size_t)MTOT * DIMC * 2);
  float*    x1     = (float*)alloc((size_t)MTOT * DIMC * 4);
  _Float16* h2     = (_Float16*)alloc((size_t)MTOT * DIMC * 2);
  _Float16* hid    = (_Float16*)alloc((size_t)MTOT * HIDDEN * 2);

  // Weights -> f16 transposed (columns of B contiguous)
  cast_transpose<<<(DIMC * 3 * DIMC) / 256, 256, 0, stream>>>(qkv_w, wqkvT, DIMC, 3 * DIMC);
  cast_transpose<<<(DIMC * DIMC) / 256, 256, 0, stream>>>(proj_w, wprojT, DIMC, DIMC);
  cast_transpose<<<(DIMC * HIDDEN) / 256, 256, 0, stream>>>(fc1_w, wfc1T, DIMC, HIDDEN);
  cast_transpose<<<(HIDDEN * DIMC) / 256, 256, 0, stream>>>(fc2_w, wfc2T, HIDDEN, DIMC);

  // LN1 -> f16
  ln_kernel<<<MTOT, 256, 0, stream>>>(x, ln1_w, ln1_b, xh);

  // QKV GEMM, scatter into head layouts (V transposed for PV WMMA)
  dim3 gq(3 * DIMC / 128, MTOT / 128);
  gemm_wmma<0><<<gq, 256, 0, stream>>>(xh, wqkvT, qkv_b, MTOT, 3 * DIMC, DIMC,
                                       nullptr, nullptr, nullptr, Qb, Kb, Vtb);

  // Attention
  attn_kernel<<<BATCH * HEADS * (SEQ / 64), 128, 0, stream>>>(Qb, Kb, Vtb, attn_o);

  // Projection + residual -> x1 (fp32)
  dim3 gp(DIMC / 128, MTOT / 128);
  gemm_wmma<1><<<gp, 256, 0, stream>>>(attn_o, wprojT, proj_b, MTOT, DIMC, DIMC,
                                       nullptr, x1, x, nullptr, nullptr, nullptr);

  // LN2 -> f16
  ln_kernel<<<MTOT, 256, 0, stream>>>(x1, ln2_w, ln2_b, h2);

  // FC1 + exact GELU -> f16 hidden
  dim3 g1(HIDDEN / 128, MTOT / 128);
  gemm_wmma<2><<<g1, 256, 0, stream>>>(h2, wfc1T, fc1_b, MTOT, HIDDEN, DIMC,
                                       hid, nullptr, nullptr, nullptr, nullptr, nullptr);

  // FC2 + residual -> output (fp32)
  dim3 g2(DIMC / 128, MTOT / 128);
  gemm_wmma<1><<<g2, 256, 0, stream>>>(hid, wfc2T, fc2_b, MTOT, DIMC, HIDDEN,
                                       nullptr, outp, x1, nullptr, nullptr, nullptr);
}
